// GQAWithRoPE_82171314307648
// MI455X (gfx1250) — compile-verified, matrix operand order checked by reading, not run on hardware
//
#include <hip/hip_runtime.h>

// ---------------- problem constants (match reference) ----------------
constexpr int B    = 2;
constexpr int T    = 2048;
constexpr int D    = 2048;
constexpr int H    = 32;
constexpr int HKV  = 8;
constexpr int DH   = 64;          // head dim
constexpr int G    = H / HKV;     // 4
constexpr int MT   = B * T;       // 4096 rows
constexpr int DKV  = HKV * DH;    // 512

typedef __bf16 bf16_t;
typedef __attribute__((ext_vector_type(16))) __bf16 v16bf;
typedef __attribute__((ext_vector_type(8)))  __bf16 v8bf;
typedef __attribute__((ext_vector_type(8)))  float  v8f;
typedef __attribute__((ext_vector_type(4)))  unsigned int u32x4;
typedef __attribute__((ext_vector_type(8)))  int i32x8;
typedef __attribute__((ext_vector_type(4)))  int i32x4;

#define NEG_INF (-3.402823466e38f)

// Load a 16x32 bf16 A-style fragment (also used for our "transposed B" layout).
// Per CDNA5 ISA: lane l -> row m = l&15; half h = l>>4;
//   vector elems 0..7  = dims [kb + h*8,        kb + h*8 + 8)
//   vector elems 8..15 = dims [kb + 16 + h*8,   kb + 16 + h*8 + 8)
__device__ __forceinline__ v16bf load_frag(const bf16_t* rowp, int kb, int h) {
  v8bf lo = *reinterpret_cast<const v8bf*>(rowp + kb + h * 8);
  v8bf hi = *reinterpret_cast<const v8bf*>(rowp + kb + 16 + h * 8);
  v16bf f;
#pragma unroll
  for (int i = 0; i < 8; ++i) { f[i] = lo[i]; f[i + 8] = hi[i]; }
  return f;
}

// ---------------- Tensor Data Mover: 2-D tile (rows x 64 bf16) -> LDS ----------------
// D# per CDNA5 ISA section 8.3/8.4. data_size=1 (2 bytes), tile_dim0=64 elems (128B),
// tile_dim1=rows, tensor_dim0_stride = row pitch (elements).
__device__ __forceinline__ void tdm_load_tile(unsigned int lds_off_bytes,
                                              const bf16_t* gaddr,
                                              int row_pitch, int rows) {
  unsigned long long ga = (unsigned long long)(uintptr_t)gaddr;
  u32x4 g0;
  g0[0] = 1u;                                            // count=1, user mode
  g0[1] = lds_off_bytes;                                 // lds_addr
  g0[2] = (unsigned int)ga;                              // global_addr[31:0]
  g0[3] = (unsigned int)((ga >> 32) & 0x01FFFFFFu) | (2u << 30);  // addr[56:32] | type=2
  i32x8 g1;
  g1[0] = (int)(1u << 16);                               // wg_mask=0, data_size=1 (2B)
  g1[1] = (int)(((unsigned)row_pitch & 0xFFFFu) << 16);  // tensor_dim0[15:0] (<<48)
  g1[2] = (int)((((unsigned)row_pitch >> 16) & 0xFFFFu)  // tensor_dim0[31:16]
                | (((unsigned)rows & 0xFFFFu) << 16));   // tensor_dim1[15:0]
  g1[3] = (int)((((unsigned)rows >> 16) & 0xFFFFu)       // tensor_dim1[31:16]
                | (64u << 16));                          // tile_dim0 = 64
  g1[4] = rows;                                          // tile_dim1 (tile_dim2=0)
  g1[5] = row_pitch;                                     // tensor_dim0_stride[31:0]
  g1[6] = 0;
  g1[7] = 0;
  i32x4 z4 = {};
#if __clang_major__ >= 23
  i32x8 z8 = {};
  __builtin_amdgcn_tensor_load_to_lds(g0, g1, z4, z4, z8, 0);
#else
  __builtin_amdgcn_tensor_load_to_lds(g0, g1, z4, z4, 0);
#endif
}

// ---------------- conversion kernels ----------------
__global__ void f32_to_bf16_kernel(const float* __restrict__ in, bf16_t* __restrict__ out, int n) {
  int i = blockIdx.x * blockDim.x + threadIdx.x;
  if (i < n) out[i] = (bf16_t)in[i];
}

// W[K][N] (f32, row major) -> Wt[N][K] (bf16)
__global__ void transpose_to_bf16_kernel(const float* __restrict__ W, bf16_t* __restrict__ Wt,
                                         int K, int N) {
  int i = blockIdx.x * blockDim.x + threadIdx.x;
  if (i < K * N) {
    int k = i / N, n = i % N;
    Wt[(size_t)n * K + k] = (bf16_t)W[i];
  }
}

// V [B,T,HKV,DH] -> Vt [B,HKV,DH,T]   (bf16 -> bf16)
__global__ void v_transpose_kernel(const bf16_t* __restrict__ V, bf16_t* __restrict__ Vt) {
  int i = blockIdx.x * blockDim.x + threadIdx.x;
  if (i >= MT * DKV) return;
  int dh = i & (DH - 1);
  int j  = i >> 6;
  int hk = j % HKV;
  int j2 = j / HKV;
  int t  = j2 % T;
  int b  = j2 / T;
  Vt[(((size_t)b * HKV + hk) * DH + dh) * T + t] = V[i];
}

// in-place RoPE on bf16 tensor laid out [B,T,nh,DH]; one thread per (b,t,h,d<32)
__global__ void rope_kernel(bf16_t* __restrict__ Q, int nh, int total) {
  int i = blockIdx.x * blockDim.x + threadIdx.x;
  if (i >= total) return;
  int d  = i & 31;
  int j  = i >> 5;
  int hh = j % nh;
  int j2 = j / nh;
  int t  = j2 % T;
  int b  = j2 / T;
  float ang = (float)t / __powf(10000.0f, (float)d / 32.0f);
  float c = __cosf(ang), s = __sinf(ang);
  bf16_t* p = Q + (((size_t)b * T + t) * nh + hh) * DH;
  float x1 = (float)p[d];
  float x2 = (float)p[d + 32];
  p[d]      = (bf16_t)(x1 * c - x2 * s);
  p[d + 32] = (bf16_t)(x2 * c + x1 * s);
}

// ---------------- TDM-staged WMMA GEMM: C[M,N] = A[M,K] * Bt[N,K]^T ----------------
// block = 256 threads (8 waves). Block tile 128x128, K-step 64, double-buffered LDS
// filled by the Tensor Data Mover (TENSORcnt). Wave w computes rows [w*16, w*16+16).
template <bool OUTF32>
__global__ __launch_bounds__(256) void gemm_tdm_kernel(const bf16_t* __restrict__ A,
                                                       const bf16_t* __restrict__ Bt,
                                                       float* __restrict__ Cf,
                                                       bf16_t* __restrict__ Cb,
                                                       int M, int N, int Kdim) {
  const int lane = threadIdx.x & 31;
  const int wave = threadIdx.x >> 5;
  const int h    = lane >> 4;
  const int ln   = lane & 15;
  const int nblocks = N / 128;
  const int mb = blockIdx.x / nblocks;
  const int nb = blockIdx.x % nblocks;

  // LDS layout (this kernel's only shared object => offset 0):
  //   A buffers: [0, 16K) and [16K, 32K); B buffers: [32K, 48K) and [48K, 64K)
  __shared__ bf16_t smem[4 * 128 * 64];
  constexpr unsigned TILE_B = 128 * 64 * sizeof(bf16_t);   // 16384 bytes

  const bf16_t* Ablk = A  + (size_t)(mb * 128) * Kdim;
  const bf16_t* Bblk = Bt + (size_t)(nb * 128) * Kdim;

  const int NS = Kdim / 64;
  // prologue: stage 0 into buffer 0
  if (wave == 0) {
    tdm_load_tile(0 * TILE_B,            Ablk, Kdim, 128);
    tdm_load_tile(2 * TILE_B,            Bblk, Kdim, 128);
  }

  v8f c[8] = {};
  for (int s = 0; s < NS; ++s) {
    const int buf = s & 1;
    if (wave == 0) {
      if (s + 1 < NS) {
        const int kb2 = (s + 1) * 64;
        tdm_load_tile((unsigned)(buf ^ 1) * TILE_B,           Ablk + kb2, Kdim, 128);
        tdm_load_tile((unsigned)(2 + (buf ^ 1)) * TILE_B,     Bblk + kb2, Kdim, 128);
        __builtin_amdgcn_s_wait_tensorcnt(2);   // stage s complete (TDM in-order per wave)
      } else {
        __builtin_amdgcn_s_wait_tensorcnt(0);
      }
    }
    __syncthreads();                            // publish LDS tile `buf`

    const bf16_t* As = smem + (size_t)buf * (128 * 64);
    const bf16_t* Bs = smem + (size_t)(2 + buf) * (128 * 64);
    const bf16_t* arow = As + (wave * 16 + ln) * 64;
    v16bf af0 = load_frag(arow, 0, h);
    v16bf af1 = load_frag(arow, 32, h);
#pragma unroll
    for (int t = 0; t < 8; ++t) {
      const bf16_t* brow = Bs + (t * 16 + ln) * 64;
      v16bf bf0 = load_frag(brow, 0, h);
      v16bf bf1 = load_frag(brow, 32, h);
      c[t] = __builtin_amdgcn_wmma_f32_16x16x32_bf16(false, af0, false, bf0, (short)0, c[t], false, false);
      c[t] = __builtin_amdgcn_wmma_f32_16x16x32_bf16(false, af1, false, bf1, (short)0, c[t], false, false);
    }
    __syncthreads();                            // tile `buf` free for overwrite
  }

#pragma unroll
  for (int t = 0; t < 8; ++t) {
#pragma unroll
    for (int r = 0; r < 8; ++r) {
      int row = mb * 128 + wave * 16 + r + 8 * h;
      int col = nb * 128 + t * 16 + ln;
      float v = c[t][r];
      if (OUTF32) Cf[(size_t)row * N + col] = v;
      else        Cb[(size_t)row * N + col] = (bf16_t)v;
    }
  }
  // cluster-wide barrier: NOP when not launched in a cluster (ClusterID==0)
  __builtin_amdgcn_s_cluster_barrier();
}

// ---------------- flash attention (causal, GQA) ----------------
// Q  : [B,T,H,DH]   bf16 (post-RoPE);  K : [B,T,HKV,DH] bf16 (post-RoPE)
// Vt : [B,HKV,DH,T] bf16;              Ctx: [B,T,H,DH]  bf16
// block = 128 threads (4 waves); each wave owns 16 query rows of one head.
__global__ __launch_bounds__(128) void attn_kernel(const bf16_t* __restrict__ Q,
                                                   const bf16_t* __restrict__ K,
                                                   const bf16_t* __restrict__ Vt,
                                                   bf16_t* __restrict__ Ctx) {
  const int lane = threadIdx.x & 31;
  const int wave = threadIdx.x >> 5;
  const int h    = lane >> 4;
  const int ln   = lane & 15;

  const int tiles = T / 64;
  int bid  = blockIdx.x;
  int b    = bid / (H * tiles);
  int rem  = bid % (H * tiles);
  int hq   = rem / tiles;
  int tile = rem % tiles;
  int qbase = tile * 64 + wave * 16;
  int hkv  = hq / G;

  __shared__ bf16_t plds[4 * 16 * 32];
  bf16_t* pw = plds + wave * (16 * 32);

  const bf16_t* qrow = Q + (((size_t)b * T + qbase + ln) * H + hq) * DH;
  v16bf qf0 = load_frag(qrow, 0, h);
  v16bf qf1 = load_frag(qrow, 32, h);

  float rm[8], rs[8];
#pragma unroll
  for (int r = 0; r < 8; ++r) { rm[r] = NEG_INF; rs[r] = 0.0f; }
  v8f o[4] = {};

  const int qlast = qbase + 15;
  for (int kb = 0; kb <= qlast; kb += 32) {
    const bf16_t* krow0 = K + (((size_t)b * T + kb + ln)      * HKV + hkv) * DH;
    const bf16_t* krow1 = K + (((size_t)b * T + kb + 16 + ln) * HKV + hkv) * DH;
    __builtin_prefetch(krow0 + 32 * HKV * DH, 0, 0);     // global_prefetch_b8 (next kb)
    v8f s0 = {}, s1 = {};
#pragma unroll
    for (int ks = 0; ks < 2; ++ks) {
      v16bf a  = ks ? qf1 : qf0;
      v16bf k0 = load_frag(krow0, ks * 32, h);
      v16bf k1 = load_frag(krow1, ks * 32, h);
      s0 = __builtin_amdgcn_wmma_f32_16x16x32_bf16(false, a, false, k0, (short)0, s0, false, false);
      s1 = __builtin_amdgcn_wmma_f32_16x16x32_bf16(false, a, false, k1, (short)0, s1, false, false);
    }

    const int kj0 = kb + ln;
    const int kj1 = kb + 16 + ln;
#pragma unroll
    for (int r = 0; r < 8; ++r) {
      const int m  = r + 8 * h;
      const int qi = qbase + m;
      float sv0 = s0[r] * 0.125f;
      float sv1 = s1[r] * 0.125f;
      if (kj0 > qi) sv0 = NEG_INF;
      if (kj1 > qi) sv1 = NEG_INF;

      float mx = fmaxf(sv0, sv1);
#pragma unroll
      for (int msk = 1; msk < 16; msk <<= 1) mx = fmaxf(mx, __shfl_xor(mx, msk, 32));
      float mnew  = fmaxf(rm[r], mx);
      float scale = __expf(rm[r] - mnew);
      float p0 = __expf(sv0 - mnew);
      float p1 = __expf(sv1 - mnew);
      float ps = p0 + p1;
#pragma unroll
      for (int msk = 1; msk < 16; msk <<= 1) ps += __shfl_xor(ps, msk, 32);
      rs[r] = rs[r] * scale + ps;
      rm[r] = mnew;
#pragma unroll
      for (int t = 0; t < 4; ++t) o[t][r] *= scale;

      pw[m * 32 + ln]      = (bf16_t)p0;
      pw[m * 32 + 16 + ln] = (bf16_t)p1;
    }

    v16bf pa;
    {
      v8bf lo = *reinterpret_cast<const v8bf*>(pw + ln * 32 + h * 8);
      v8bf hi = *reinterpret_cast<const v8bf*>(pw + ln * 32 + 16 + h * 8);
#pragma unroll
      for (int i = 0; i < 8; ++i) { pa[i] = lo[i]; pa[i + 8] = hi[i]; }
    }

#pragma unroll
    for (int t = 0; t < 4; ++t) {
      const bf16_t* vrow = Vt + (((size_t)b * HKV + hkv) * DH + t * 16 + ln) * T;
      v16bf vf = load_frag(vrow, kb, h);
      o[t] = __builtin_amdgcn_wmma_f32_16x16x32_bf16(false, pa, false, vf, (short)0, o[t], false, false);
    }
  }

#pragma unroll
  for (int r = 0; r < 8; ++r) {
    float inv = 1.0f / rs[r];
    int m = r + 8 * h;
    size_t base = (((size_t)b * T + qbase + m) * H + hq) * DH;
#pragma unroll
    for (int t = 0; t < 4; ++t) Ctx[base + t * 16 + ln] = (bf16_t)(o[t][r] * inv);
  }
}

// ---------------- host-side orchestration ----------------
extern "C" void kernel_launch(void* const* d_in, const int* in_sizes, int n_in,
                              void* d_out, int out_size, void* d_ws, size_t ws_size,
                              hipStream_t stream) {
  (void)in_sizes; (void)n_in; (void)out_size; (void)ws_size;
  const float* x  = (const float*)d_in[0];
  const float* Wq = (const float*)d_in[1];
  const float* Wk = (const float*)d_in[2];
  const float* Wv = (const float*)d_in[3];
  const float* Wo = (const float*)d_in[4];
  float* out = (float*)d_out;

  char* p = (char*)d_ws;
  auto carve = [&](size_t elems) -> bf16_t* {
    bf16_t* r = (bf16_t*)p;
    size_t bytes = (elems * sizeof(bf16_t) + 255) & ~(size_t)255;
    p += bytes;
    return r;
  };
  bf16_t* xb   = carve((size_t)MT * D);
  bf16_t* wqt  = carve((size_t)D * D);
  bf16_t* wkt  = carve((size_t)DKV * D);
  bf16_t* wvt  = carve((size_t)DKV * D);
  bf16_t* wot  = carve((size_t)D * D);
  bf16_t* qb   = carve((size_t)MT * D);
  bf16_t* kbuf = carve((size_t)MT * DKV);
  bf16_t* vbuf = carve((size_t)MT * DKV);
  bf16_t* vtb  = carve((size_t)MT * DKV);
  bf16_t* ctxb = carve((size_t)MT * D);

  const int THR = 256;
  f32_to_bf16_kernel<<<(MT * D) / THR, THR, 0, stream>>>(x, xb, MT * D);
  transpose_to_bf16_kernel<<<(D * D) / THR, THR, 0, stream>>>(Wq, wqt, D, D);
  transpose_to_bf16_kernel<<<(D * DKV) / THR, THR, 0, stream>>>(Wk, wkt, D, DKV);
  transpose_to_bf16_kernel<<<(D * DKV) / THR, THR, 0, stream>>>(Wv, wvt, D, DKV);
  transpose_to_bf16_kernel<<<(D * D) / THR, THR, 0, stream>>>(Wo, wot, D, D);

  gemm_tdm_kernel<false><<<(MT / 128) * (D / 128),   256, 0, stream>>>(xb, wqt, nullptr, qb,   MT, D,   D);
  gemm_tdm_kernel<false><<<(MT / 128) * (DKV / 128), 256, 0, stream>>>(xb, wkt, nullptr, kbuf, MT, DKV, D);
  gemm_tdm_kernel<false><<<(MT / 128) * (DKV / 128), 256, 0, stream>>>(xb, wvt, nullptr, vbuf, MT, DKV, D);

  rope_kernel<<<(MT * H * 32) / THR, THR, 0, stream>>>(qb, H, MT * H * 32);
  rope_kernel<<<(MT * HKV * 32) / THR, THR, 0, stream>>>(kbuf, HKV, MT * HKV * 32);

  v_transpose_kernel<<<(MT * DKV) / THR, THR, 0, stream>>>(vbuf, vtb);

  attn_kernel<<<B * H * (T / 64), 128, 0, stream>>>(qb, kbuf, vtb, ctxb);

  gemm_tdm_kernel<true><<<(MT / 128) * (D / 128), 256, 0, stream>>>(ctxb, wot, out, nullptr, MT, D, D);
}
